// IntVOS_27015344292356
// MI455X (gfx1250) — matrix operand order repaired and checked by reading, not compiled, not run on hardware
//
#include <hip/hip_runtime.h>
#include <hip/hip_bf16.h>

typedef __attribute__((ext_vector_type(16))) _Float16 v16h;
typedef __attribute__((ext_vector_type(8)))  float    v8f;

// Problem geometry (fixed by the reference)
constexpr int HF = 96, WF = 160, C = 100;      // full-res
constexpr int HL = 48, WL = 80, CP = 128;      // low-res, padded channels
constexpr int MRAD = 9, KD = 19, NK = KD * KD; // 361 offsets
constexpr int YH = HL + 2 * MRAD;              // 66  (h' = idx-9)
constexpr int YW = WL + 32;                    // 112 (w' = idx-16, tile-aligned)

// ---------------- Stage 1a: downsample + f16 convert query -> X ----------------
__global__ void k_ds_x(const float* __restrict__ q, _Float16* __restrict__ X) {
    int idx = blockIdx.x * blockDim.x + threadIdx.x;
    if (idx >= HL * WL * CP) return;
    int c = idx & (CP - 1);
    int w = (idx >> 7) % WL;
    int h = idx / (CP * WL);
    float v = 0.f;
    if (c < C) {
        const float* p = q + ((2 * h) * WF + 2 * w) * C + c;
        v = 0.25f * (p[0] + p[C] + p[WF * C] + p[WF * C + C]);
    }
    X[idx] = (_Float16)v;
}

// ---------------- Stage 1b: downsample + f16 convert prev -> padded Y ----------------
__global__ void k_ds_y(const float* __restrict__ prev, _Float16* __restrict__ Y) {
    int idx = blockIdx.x * blockDim.x + threadIdx.x;
    if (idx >= YH * YW * CP) return;
    int c  = idx & (CP - 1);
    int wp = (idx >> 7) % YW;
    int hp = idx / (CP * YW);
    int h = hp - MRAD, w = wp - 16;
    float v = 0.f;
    if (c < C && h >= 0 && h < HL && w >= 0 && w < WL) {
        const float* p = prev + ((2 * h) * WF + 2 * w) * C + c;
        v = 0.25f * (p[0] + p[C] + p[WF * C] + p[WF * C + C]);
    }
    Y[idx] = (_Float16)v;
}

// ---------------- Stage 2: fp32 squared norms (pad -> 1e30 reproduces PAD_VAL) ----------------
__global__ void k_norms(const float* __restrict__ q, const float* __restrict__ prev,
                        float* __restrict__ xsq, float* __restrict__ ysq) {
    int i = blockIdx.x * blockDim.x + threadIdx.x;
    if (i < HL * WL) {
        int h = i / WL, w = i % WL;
        const float* p = q + ((2 * h) * WF + 2 * w) * C;
        float s = 0.f;
        for (int c = 0; c < C; ++c) {
            float a = 0.25f * (p[c] + p[c + C] + p[c + WF * C] + p[c + WF * C + C]);
            s += a * a;
        }
        xsq[i] = s;
    } else if (i < HL * WL + YH * YW) {
        int i2 = i - HL * WL;
        int hp = i2 / YW, wp = i2 % YW;
        int h = hp - MRAD, w = wp - 16;
        float s = 1e30f;
        if (h >= 0 && h < HL && w >= 0 && w < WL) {
            const float* p = prev + ((2 * h) * WF + 2 * w) * C;
            s = 0.f;
            for (int c = 0; c < C; ++c) {
                float a = 0.25f * (p[c] + p[c + C] + p[c + WF * C] + p[c + WF * C + C]);
                s += a * a;
            }
        }
        ysq[i2] = s;
    }
}

// ---------------- Stage 3: banded correlation via v_wmma_f32_16x16x32_f16 ----------------
// One wave32 per (h, dy, w-tile). Three independent accumulator chains (jj=0,1,2)
// are interleaved so consecutive WMMA issues have no D->C dependency (no hazard NOPs).
__global__ void __launch_bounds__(32) k_corr(const _Float16* __restrict__ X,
                                             const _Float16* __restrict__ Y,
                                             const float* __restrict__ xsq,
                                             const float* __restrict__ ysq,
                                             float* __restrict__ dbuf) {
    int bid = blockIdx.x;
    int h  = bid % HL;
    int dy = (bid / HL) % KD;
    int it = bid / (HL * KD);          // 0..4, w-tile of X
    int lane = threadIdx.x;
    int lm = lane & 15, half = lane >> 4;
    int hp = h + dy;                   // padded row of Y (= h' + 9)

    // ---- A fragments: X tile rows, ISA 16-bit A layout ----
    // lanes 0-15: M=lane, elems 0..7 -> K = 8*half + e ; elems 8..15 -> K = 16 + 8*half + e
    v16h A[4];
    {
        int wx = 16 * it + lm;
        const _Float16* abase = X + ((size_t)(h * WL + wx) << 7);
#pragma unroll
        for (int s = 0; s < 4; ++s) {
            const _Float16* ap = abase + 32 * s + 8 * half;
#pragma unroll
            for (int e = 0; e < 8; ++e) {
                A[s][e]     = ap[e];
                A[s][8 + e] = ap[16 + e];
            }
        }
    }

    float xn[8];
#pragma unroll
    for (int r = 0; r < 8; ++r) xn[r] = xsq[h * WL + 16 * it + r + 8 * half];

    // ---- three interleaved accumulator chains ----
    // B column for chain jj lives at padded col wp = 16*(it+jj) + lm
    // -> element offset from chain-0 base is jj * 16 * CP = jj * 2048
    const _Float16* bbase = Y + ((size_t)(hp * YW + 16 * it + lm) << 7);
    v8f acc[3] = {v8f{}, v8f{}, v8f{}};
#pragma unroll
    for (int s = 0; s < 4; ++s) {
        // ISA 16-bit B layout: lane N = lm, elems 0..7 -> K = 32*s + 8*half + e ; elems 8..15 -> K = 32*s + 16 + 8*half + (e-8)
        const _Float16* bp = bbase + 32 * s + 8 * half;
        v16h B0, B1, B2;
#pragma unroll
        for (int e = 0; e < 16; ++e) {
            int ke = (e & 8) ? e + 8 : e;
            B0[e] = bp[ke];
            B1[e] = bp[ke + 2048];
            B2[e] = bp[ke + 4096];
        }
        acc[0] = __builtin_amdgcn_wmma_f32_16x16x32_f16(false, A[s], false, B0,
                                                        (short)0, acc[0], false, false);
        acc[1] = __builtin_amdgcn_wmma_f32_16x16x32_f16(false, A[s], false, B1,
                                                        (short)0, acc[1], false, false);
        acc[2] = __builtin_amdgcn_wmma_f32_16x16x32_f16(false, A[s], false, B2,
                                                        (short)0, acc[2], false, false);
    }

    // ---- epilogue: lane owns C elements (M = r + 8*half, N = lm) ----
#pragma unroll
    for (int jj = 0; jj < 3; ++jj) {
        int wp    = 16 * (it + jj) + lm;          // padded col, 0..111
        float yn  = ysq[hp * YW + wp];
        int   wjg = wp - 16;                      // global w'
#pragma unroll
        for (int r = 0; r < 8; ++r) {
            int wi = 16 * it + r + 8 * half;
            int dx = wjg - wi + MRAD;
            if (dx >= 0 && dx < KD) {
                float d = xn[r] + yn - 2.0f * acc[jj][r];
                // (sigmoid(d)-0.5)*2 == (1-e)/(1+e), e = exp(-d); single v_rcp_f32
                float e = __expf(-d);
                float v = (1.0f - e) * __builtin_amdgcn_rcpf(1.0f + e);
                dbuf[(size_t)(h * WL + wi) * NK + dy * KD + dx] = v;
            }
        }
    }
}

// ---------------- Stage 4: bilinear upsample + label-masked min over offsets ----------------
__global__ void k_maskmin(const float* __restrict__ dbuf, const int* __restrict__ labels,
                          const int* __restrict__ gt, int nobj, float* __restrict__ out) {
    int pix = blockIdx.x * blockDim.x + threadIdx.x;
    if (pix >= HF * WF) return;
    int Hi = pix / WF, Wi = pix % WF;

    // align_corners bilinear taps (2 per axis)
    float ph = Hi * ((float)(HL - 1) / (float)(HF - 1));
    int lh = (int)floorf(ph); if (lh > HL - 2) lh = HL - 2; if (lh < 0) lh = 0;
    float fh = ph - (float)lh;
    float pw = Wi * ((float)(WL - 1) / (float)(WF - 1));
    int lw = (int)floorf(pw); if (lw > WL - 2) lw = WL - 2; if (lw < 0) lw = 0;
    float fw = pw - (float)lw;
    float w00 = (1.f - fh) * (1.f - fw), w01 = (1.f - fh) * fw;
    float w10 = fh * (1.f - fw),         w11 = fh * fw;
    const float* p00 = dbuf + (size_t)(lh * WL + lw) * NK;
    const float* p01 = p00 + NK;
    const float* p10 = p00 + (size_t)WL * NK;
    const float* p11 = p10 + NK;

    float mins[16];
    int   g[16];
    for (int n = 0; n < nobj; ++n) { mins[n] = 1.0f; g[n] = gt[n]; }

    for (int dy = 0; dy < KD; ++dy) {
        int hh = Hi + dy - MRAD;
        if (hh < 0 || hh >= HF) continue;
        const int* lrow = labels + hh * WF;
        for (int dx = 0; dx < KD; ++dx) {
            int ww = Wi + dx - MRAD;
            if (ww < 0 || ww >= WF) continue;
            int lab = lrow[ww];
            int k = dy * KD + dx;
            float v = 0.f; bool got = false;
            for (int n = 0; n < nobj; ++n) {
                if (lab == g[n]) {
                    if (!got) {
                        v = w00 * p00[k] + w01 * p01[k] + w10 * p10[k] + w11 * p11[k];
                        got = true;
                    }
                    if (v < mins[n]) mins[n] = v;
                }
            }
        }
    }
    for (int n = 0; n < nobj; ++n) out[(size_t)pix * nobj + n] = mins[n];
}

extern "C" void kernel_launch(void* const* d_in, const int* in_sizes, int n_in,
                              void* d_out, int out_size, void* d_ws, size_t ws_size,
                              hipStream_t stream) {
    const float* prev   = (const float*)d_in[0];
    const float* query  = (const float*)d_in[1];
    const int*   labels = (const int*)d_in[2];
    const int*   gt     = (const int*)d_in[3];
    int nobj = in_sizes[3];
    if (nobj > 16) nobj = 16;

    char* ws = (char*)d_ws;
    size_t off = 0;
    auto alloc = [&](size_t bytes) -> void* {
        void* p = ws + off;
        off += (bytes + 255) & ~(size_t)255;
        return p;
    };
    _Float16* X    = (_Float16*)alloc((size_t)HL * WL * CP * 2);
    _Float16* Y    = (_Float16*)alloc((size_t)YH * YW * CP * 2);
    float*    xsq  = (float*)alloc((size_t)HL * WL * 4);
    float*    ysq  = (float*)alloc((size_t)YH * YW * 4);
    float*    dbuf = (float*)alloc((size_t)HL * WL * NK * 4);
    (void)ws_size;

    k_ds_x<<<(HL * WL * CP + 255) / 256, 256, 0, stream>>>(query, X);
    k_ds_y<<<(YH * YW * CP + 255) / 256, 256, 0, stream>>>(prev, Y);
    k_norms<<<(HL * WL + YH * YW + 255) / 256, 256, 0, stream>>>(query, prev, xsq, ysq);
    k_corr<<<HL * KD * (WL / 16), 32, 0, stream>>>(X, Y, xsq, ysq, dbuf);
    k_maskmin<<<(HF * WF + 255) / 256, 256, 0, stream>>>(dbuf, labels, gt, nobj, (float*)d_out);
}
